// MultiHeadAttentionKernelClassical_65481071399302
// MI455X (gfx1250) — matrix operand order repaired and checked by reading, not compiled, not run on hardware
//
#include <hip/hip_runtime.h>
#include <hip/hip_bf16.h>

// ---------------------------------------------------------------------------
// RBF-kernel MHA for MI455X (gfx1250, wave32).
// Compute-bound (~69 GFLOP vs ~50MB traffic) -> all matmuls via
// v_wmma_f32_16x16x32_bf16. GEMMs use 64x64 register blocking per wave
// (16 WMMAs per K-step, ~32 FLOP/byte of fragment traffic).
// __launch_bounds__(256,1) grants the full VGPR budget (no accumulator
// spills); output-layout mode is a template param for a branch-free epilogue.
// ---------------------------------------------------------------------------

typedef __attribute__((ext_vector_type(16))) __bf16 v16bf;
typedef __attribute__((ext_vector_type(8)))  float  v8f;

#define B_   2
#define N_   2048
#define E_   1024
#define H_   16
#define DK_  64
#define GM   4096   // B_*N_
#define GK   1024
#define GN   1024

__device__ __forceinline__ v8f wmma_bf16(v16bf a, v16bf b, v8f c) {
    return __builtin_amdgcn_wmma_f32_16x16x32_bf16(
        false, a, false, b, (short)0, c, false, false);
}

// A fragment, 16x32 bf16, source row-major with leading dim ld.
// ISA layout: lanes 0-15 -> M=lane, K in {0..7, 16..23}; lanes 16-31 ->
// M=lane-16, K in {8..15, 24..31}. Two b128 loads per lane.
__device__ __forceinline__ v16bf frag_a_load(const __bf16* base, int row0, int ld,
                                             int k0, int lane) {
    int r  = lane & 15;
    int kb = (lane < 16) ? 0 : 8;
    const __bf16* p = base + (size_t)(row0 + r) * ld + k0 + kb;
    union { v16bf v; uint4 q[2]; } u;
    u.q[0] = *(const uint4*)(p);
    u.q[1] = *(const uint4*)(p + 16);
    return u.v;
}

// B fragment, 32x16 bf16 (KxN). Source holds matrix columns as contiguous rows:
// src[n][k] = B[k][n] (leading dim ld). ISA layout: lanes 0-15 -> N=lane,
// K in {0..7, 16..23}; lanes 16-31 -> N=lane-16, K in {8..15, 24..31}. Two b128 loads.
__device__ __forceinline__ v16bf frag_b_load(const __bf16* base, int row0, int ld,
                                             int k0, int lane) {
    int n  = lane & 15;
    int kb = (lane < 16) ? 0 : 8;
    const __bf16* p = base + (size_t)(row0 + n) * ld + k0 + kb;
    union { v16bf v; uint4 q[2]; } u;
    u.q[0] = *(const uint4*)(p);
    u.q[1] = *(const uint4*)(p + 16);
    return u.v;
}

// ---------------------------------------------------------------------------
__global__ void cvt_f32_bf16(const float* __restrict__ src, __bf16* __restrict__ dst,
                             int n) {
    int i = blockIdx.x * blockDim.x + threadIdx.x;
    if (i < n) dst[i] = (__bf16)src[i];
}

// Row sum of squares over DK_=64 bf16 elements -> f32.
__global__ void rowsumsq(const __bf16* __restrict__ X, float* __restrict__ out,
                         int rows) {
    int r = blockIdx.x * blockDim.x + threadIdx.x;
    if (r >= rows) return;
    const __bf16* p = X + (size_t)r * DK_;
    float s = 0.0f;
#pragma unroll
    for (int d = 0; d < DK_; ++d) { float v = (float)p[d]; s += v * v; }
    out[r] = s;
}

// C = A(4096x1024, bf16 row-major) x W^T (W: 1024x1024 bf16 row-major).
// One wave computes a 64x64 output tile: 4x4 C fragments, 16 WMMAs per K-step.
// MODE 0: store bf16 head-major      out[((b*16+h)*2048+tok)*64 + d]
// MODE 1: store bf16 head-transposed out[((b*16+h)*64+d)*2048 + tok]  (for V)
// MODE 2: store f32 row-major        out[m*1024 + ncol]               (final)
template <int MODE>
__global__ __launch_bounds__(256, 1)
void gemm16(const __bf16* __restrict__ A, const __bf16* __restrict__ W,
            void* __restrict__ Out) {
    int lane = threadIdx.x & 31;
    int wid  = blockIdx.x * (blockDim.x >> 5) + (threadIdx.x >> 5);
    const int ntn = GN / 64;          // 16 tiles of 64 in N
    int m0 = (wid / ntn) * 64;        // 0..4032
    int n0 = (wid % ntn) * 64;        // 0..960

    v8f acc[4][4] = {};
    for (int k0 = 0; k0 < GK; k0 += 32) {
        // Prefetch next K-panel (global_prefetch_b8).
        if (k0 + 32 < GK) {
            __builtin_prefetch(A + (size_t)(m0 + (lane & 15)) * GK + k0 + 32, 0, 1);
            __builtin_prefetch(W + (size_t)(n0 + (lane & 15)) * GK + k0 + 32, 0, 1);
        }
        v16bf a[4], b[4];
#pragma unroll
        for (int i = 0; i < 4; ++i) a[i] = frag_a_load(A, m0 + 16 * i, GK, k0, lane);
#pragma unroll
        for (int j = 0; j < 4; ++j) b[j] = frag_b_load(W, n0 + 16 * j, GK, k0, lane);
#pragma unroll
        for (int i = 0; i < 4; ++i)
#pragma unroll
            for (int j = 0; j < 4; ++j)
                acc[i][j] = wmma_bf16(a[i], b[j], acc[i][j]);
    }

    int half8 = (lane < 16) ? 0 : 8;
    int n     = lane & 15;
#pragma unroll
    for (int i = 0; i < 4; ++i) {
#pragma unroll
        for (int j = 0; j < 4; ++j) {
            int ncol = n0 + 16 * j + n;
#pragma unroll
            for (int r = 0; r < 8; ++r) {
                int   m   = m0 + 16 * i + r + half8;
                float val = acc[i][j][r];
                if (MODE == 2) {
                    ((float*)Out)[(size_t)m * GN + ncol] = val;
                } else {
                    int bb  = m >> 11;        // / N_
                    int tok = m & (N_ - 1);
                    int h   = ncol >> 6;
                    int d   = ncol & 63;
                    if (MODE == 0)
                        ((__bf16*)Out)[(((size_t)(bb * H_ + h) * N_ + tok) << 6) + d] = (__bf16)val;
                    else
                        ((__bf16*)Out)[((size_t)(bb * H_ + h) * DK_ + d) * N_ + tok] = (__bf16)val;
                }
            }
        }
    }
}

// One wave per 16-query tile. Q,K head-major (bh, n, 64); Vt head-transposed
// (bh, 64, n). Single-pass softmax (similarity bounded in [0,1]).
__global__ __launch_bounds__(256, 1)
void attn_kernel(const __bf16* __restrict__ Q, const __bf16* __restrict__ K,
                 const __bf16* __restrict__ Vt,
                 const float* __restrict__ qsq, const float* __restrict__ ksq,
                 const int* __restrict__ mask, __bf16* __restrict__ O) {
    __shared__ __align__(16) __bf16 plds[8][16][32];  // wave-private P tiles
    __shared__ float denl[8][32][8];                  // denominator reduction

    int lane = threadIdx.x & 31;
    int w    = threadIdx.x >> 5;
    int wid  = blockIdx.x * 8 + w;
    int bh   = wid >> 7;              // / (N_/16)
    int qt   = wid & 127;
    int bb   = bh >> 4;
    int h    = bh & 15;
    int q0   = qt * 16;

    const __bf16* Qbh = Q  + (size_t)bh * N_ * DK_;
    const __bf16* Kbh = K  + (size_t)bh * N_ * DK_;
    const __bf16* Vbh = Vt + (size_t)bh * DK_ * N_;
    const float* qsqb = qsq + (size_t)bh * N_;
    const float* ksqb = ksq + (size_t)bh * N_;
    const int*   mb   = mask + (size_t)bb * N_;

    v16bf qa0 = frag_a_load(Qbh, q0, DK_, 0, lane);
    v16bf qa1 = frag_a_load(Qbh, q0, DK_, 32, lane);
    int   half8 = (lane < 16) ? 0 : 8;
    int   n     = lane & 15;
    float qv[8];
#pragma unroll
    for (int r = 0; r < 8; ++r) qv[r] = qsqb[q0 + r + half8];

    v8f acc0 = {}, acc1 = {}, acc2 = {}, acc3 = {};
    float den[8] = {0, 0, 0, 0, 0, 0, 0, 0};

    for (int kt = 0; kt < N_; kt += 32) {
#pragma unroll
        for (int sub = 0; sub < 2; ++sub) {
            int key0 = kt + sub * 16;
            v16bf kb0 = frag_b_load(Kbh, key0, DK_, 0, lane);
            v16bf kb1 = frag_b_load(Kbh, key0, DK_, 32, lane);
            v8f s = {};
            s = wmma_bf16(qa0, kb0, s);
            s = wmma_bf16(qa1, kb1, s);
            float kv = ksqb[key0 + n];
            int   mv = mb[key0 + n];
#pragma unroll
            for (int r = 0; r < 8; ++r) {
                float sq  = fmaxf(qv[r] + kv - 2.0f * s[r], 0.0f);
                float sim = __expf(-sq);                     // GAMMA = 1
                float p   = (mv != 0) ? __expf(sim) : 0.0f;  // softmax numerator
                den[r] += p;
                plds[w][r + half8][sub * 16 + n] = (__bf16)p;
            }
        }
        // Reload P (16x32) in A-fragment layout from LDS (same-wave DS ordering;
        // compiler inserts s_wait_dscnt).
        v16bf pa  = frag_a_load(&plds[w][0][0], 0, 32, 0, lane);
        v16bf vb0 = frag_b_load(Vbh, 0,  N_, kt, lane);
        v16bf vb1 = frag_b_load(Vbh, 16, N_, kt, lane);
        v16bf vb2 = frag_b_load(Vbh, 32, N_, kt, lane);
        v16bf vb3 = frag_b_load(Vbh, 48, N_, kt, lane);
        acc0 = wmma_bf16(pa, vb0, acc0);
        acc1 = wmma_bf16(pa, vb1, acc1);
        acc2 = wmma_bf16(pa, vb2, acc2);
        acc3 = wmma_bf16(pa, vb3, acc3);
    }

    // Denominator: row m's partials live in one lane-half; reduce via LDS.
#pragma unroll
    for (int r = 0; r < 8; ++r) denl[w][lane][r] = den[r];
    int hb = (lane < 16) ? 0 : 16;
    float dsum[8];
#pragma unroll
    for (int r = 0; r < 8; ++r) {
        float s = 0.0f;
        for (int t = 0; t < 16; ++t) s += denl[w][hb + t][r];
        dsum[r] = s;
    }

    // Merge heads: O is (B, N, E) bf16 row-major.
#pragma unroll
    for (int r = 0; r < 8; ++r) {
        size_t row = (size_t)(bb * N_ + q0 + r + half8) * E_ + (size_t)h * DK_;
        float inv = 1.0f / dsum[r];
        O[row + 0  + n] = (__bf16)(acc0[r] * inv);
        O[row + 16 + n] = (__bf16)(acc1[r] * inv);
        O[row + 32 + n] = (__bf16)(acc2[r] * inv);
        O[row + 48 + n] = (__bf16)(acc3[r] * inv);
    }
}

// ---------------------------------------------------------------------------
extern "C" void kernel_launch(void* const* d_in, const int* in_sizes, int n_in,
                              void* d_out, int out_size, void* d_ws, size_t ws_size,
                              hipStream_t stream) {
    const float* x  = (const float*)d_in[0];
    const float* Wq = (const float*)d_in[1];
    const float* Wk = (const float*)d_in[2];
    const float* Wv = (const float*)d_in[3];
    const float* Wo = (const float*)d_in[4];
    const int* mask = (const int*)d_in[5];

    char* ws = (char*)d_ws;
    __bf16* xb  = (__bf16*)ws; ws += (size_t)GM * GK * 2;        // 8 MB
    __bf16* wqb = (__bf16*)ws; ws += (size_t)GN * GK * 2;        // 2 MB
    __bf16* wkb = (__bf16*)ws; ws += (size_t)GN * GK * 2;
    __bf16* wvb = (__bf16*)ws; ws += (size_t)GN * GK * 2;
    __bf16* wob = (__bf16*)ws; ws += (size_t)GN * GK * 2;
    __bf16* Qb  = (__bf16*)ws; ws += (size_t)GM * GK * 2;        // (b,h,n,d)
    __bf16* Kb  = (__bf16*)ws; ws += (size_t)GM * GK * 2;
    __bf16* Vtb = (__bf16*)ws; ws += (size_t)GM * GK * 2;        // (b,h,d,n)
    float*  qsqp = (float*)ws; ws += (size_t)B_ * H_ * N_ * 4;
    float*  ksqp = (float*)ws; ws += (size_t)B_ * H_ * N_ * 4;
    __bf16* aob = (__bf16*)ws; ws += (size_t)GM * GK * 2;        // (b,n,e)

    const int xElems = GM * GK;      // 4,194,304
    const int wElems = GN * GK;      // 1,048,576
    cvt_f32_bf16<<<(xElems + 255) / 256, 256, 0, stream>>>(x,  xb,  xElems);
    cvt_f32_bf16<<<(wElems + 255) / 256, 256, 0, stream>>>(Wq, wqb, wElems);
    cvt_f32_bf16<<<(wElems + 255) / 256, 256, 0, stream>>>(Wk, wkb, wElems);
    cvt_f32_bf16<<<(wElems + 255) / 256, 256, 0, stream>>>(Wv, wvb, wElems);
    cvt_f32_bf16<<<(wElems + 255) / 256, 256, 0, stream>>>(Wo, wob, wElems);

    // (GM/64)*(GN/64) = 1024 wave-tiles, 8 waves/block -> 128 blocks.
    gemm16<0><<<128, 256, 0, stream>>>(xb, wqb, (void*)Qb);
    gemm16<0><<<128, 256, 0, stream>>>(xb, wkb, (void*)Kb);
    gemm16<1><<<128, 256, 0, stream>>>(xb, wvb, (void*)Vtb);

    rowsumsq<<<(B_ * H_ * N_) / 256, 256, 0, stream>>>(Qb, qsqp, B_ * H_ * N_);
    rowsumsq<<<(B_ * H_ * N_) / 256, 256, 0, stream>>>(Kb, ksqp, B_ * H_ * N_);

    // B_*H_*(N_/16) = 4096 waves -> 512 blocks of 8 waves.
    attn_kernel<<<512, 256, 0, stream>>>(Qb, Kb, Vtb, qsqp, ksqp, mask, aob);

    gemm16<2><<<128, 256, 0, stream>>>(aob, wob, d_out);
}
